// RNN_13082470384272
// MI455X (gfx1250) — matrix operand order repaired and checked by reading, not compiled, not run on hardware
//
#include <hip/hip_runtime.h>

// ---- problem constants (match reference) ----
static constexpr int kBS  = 512;
static constexpr int kT   = 100;
static constexpr int kND  = 10;
static constexpr int kDS  = 6;
static constexpr int kINS = 256;
static constexpr int kHID = 256;
static constexpr int kCAT = 2816;   // kINS*kND + kHID
static constexpr int kG4  = 1024;   // 4*kHID
static constexpr int kGK  = 512;    // LSTM gemm K: [inp(256) | h(256)]

typedef __attribute__((ext_vector_type(16))) __bf16 v16bf;
typedef __attribute__((ext_vector_type(8)))  float  v8f;
typedef __attribute__((ext_vector_type(4)))  unsigned int v4u;

union Frag {
  v16bf v;
  struct { v4u lo; v4u hi; } u;
};

__device__ __forceinline__ float sigmoidf_(float x) {
  return 1.0f / (1.0f + __expf(-x));
}

// D = epilogue(A @ B^T + bias)
//   A: bf16 [M x K] row-major (lda elements), M multiple of 128 (grid-exact)
//   B: bf16 [N x K] row-major (ldb elements)
// Block tile 128(M) x 128(N), 8 waves, wave tile 32x64.
// K-tiles (128x32 of A and B) staged in LDS via async global->LDS copies,
// TRIPLE buffered on ASYNCcnt: one barrier per K-block.
// E == 0 : f32 store, +bias                   (LSTM gates)
// E == 1 : relu -> bf16 store, +bias          (z, nxt)
// E == 2 : sigmoid -> f32 store, cols<nvalid  (attention)
template <int E>
__global__ __launch_bounds__(256)
void gemm_bf16_wmma(const __bf16* __restrict__ A, int lda,
                    const __bf16* __restrict__ B, int ldb,
                    const float* __restrict__ bias,
                    void* __restrict__ D, int ldd,
                    int M, int N, int K, int nvalid)
{
  constexpr int MT = 2;             // 2 M-tiles (32 rows) per wave
  constexpr int NT = 4;             // 4 N-tiles (64 cols) per wave
  constexpr int LS = 40;            // LDS row stride: 32 + 8 pad elems (80B)
  __shared__ __bf16 sA[3][128 * LS];
  __shared__ __bf16 sB[3][128 * LS];

  const int tid  = threadIdx.x;
  const int lane = tid & 31;
  const int wave = tid >> 5;
  const int m0b  = blockIdx.x * 128;
  const int n0b  = blockIdx.y * 128;

  // copy one 16B chunk of the A tile and one of the B tile (2 async loads).
  // chunk c in [0,512): row = c>>2, 16B sub-chunk = c&3.
  auto copy_chunk = [&](int buf, int k0, int c) {
    const int row = c >> 2;
    const int o   = (c & 3) << 3;        // 0,8,16,24 elements
    {
      const __bf16* g = A + (size_t)(m0b + row) * lda + k0 + o;
      unsigned int l = (unsigned int)(uintptr_t)&sA[buf][row * LS + o];
      asm volatile("global_load_async_to_lds_b128 %0, %1, off"
                   :: "v"(l), "v"((unsigned long long)(uintptr_t)g)
                   : "memory");
    }
    {
      int rg = n0b + row;
      rg = rg > N - 1 ? N - 1 : rg;      // clamp, branch-free
      const __bf16* g = B + (size_t)rg * ldb + k0 + o;
      unsigned int l = (unsigned int)(uintptr_t)&sB[buf][row * LS + o];
      asm volatile("global_load_async_to_lds_b128 %0, %1, off"
                   :: "v"(l), "v"((unsigned long long)(uintptr_t)g)
                   : "memory");
    }
  };

  const v8f vzero = {0.f, 0.f, 0.f, 0.f, 0.f, 0.f, 0.f, 0.f};
  v8f acc[MT][NT];
#pragma unroll
  for (int mi = 0; mi < MT; ++mi)
#pragma unroll
    for (int j = 0; j < NT; ++j) acc[mi][j] = vzero;

  // CDNA5 bf16 fragment layouts:
  //  A (16x32): lane L -> row (L&15); VGPR0..3 = K [kh,kh+8), VGPR4..7 = K
  //             [16+kh,16+kh+8), kh=(L>=16)*8
  //  B (32x16): lane L -> column (L&15), same K layout as A (kb=(L>=16)*8)
  const int ar = lane & 15;
  const int ak = (lane >> 4) << 3;   // 0 or 8
  const int bn = lane & 15;
  const int bk = (lane >> 4) << 3;   // 0 or 8
  const int wm = (wave & 3) * (MT * 16);   // wave's M offset in block tile
  const int wn = (wave >> 2) * (NT * 16);  // wave's N offset in block tile

  auto compute = [&](int buf) {
    Frag fa[MT];
#pragma unroll
    for (int mi = 0; mi < MT; ++mi) {
      const __bf16* p = &sA[buf][(wm + mi * 16 + ar) * LS + ak];
      fa[mi].u.lo = *(const v4u*)p;
      fa[mi].u.hi = *(const v4u*)(p + 16);
    }
#pragma unroll
    for (int j = 0; j < NT; ++j) {
      Frag fb;
      const __bf16* p = &sB[buf][(wn + j * 16 + bn) * LS + bk];
      fb.u.lo = *(const v4u*)p;
      fb.u.hi = *(const v4u*)(p + 16);
#pragma unroll
      for (int mi = 0; mi < MT; ++mi)
        acc[mi][j] = __builtin_amdgcn_wmma_f32_16x16x32_bf16(
            false, fa[mi].v, false, fb.v, (short)0, acc[mi][j], false, false);
    }
  };

  // prologue: stage 0
  copy_chunk(0, 0, tid);
  copy_chunk(0, 0, tid + 256);

  // steady state: branch-free body, one barrier per K-block (triple buffer)
  int s = 0;
  for (int k0 = 0; k0 + 32 < K; k0 += 32) {
    const int nb = (s == 2) ? 0 : s + 1;
    copy_chunk(nb, k0 + 32, tid);
    copy_chunk(nb, k0 + 32, tid + 256);
    asm volatile("s_wait_asynccnt 0x4" ::: "memory");  // stage s landed (own)
    __syncthreads();                                    // landed for all waves
    compute(s);
    s = nb;
  }
  // tail: last K-block
  asm volatile("s_wait_asynccnt 0x0" ::: "memory");
  __syncthreads();
  compute(s);

  // C/D layout: VGPR i holds row (L>=16)*8 + i, column (L&15) of the 16x16 tile
  const int col = lane & 15;
  const int rb  = (lane >> 4) << 3;
#pragma unroll
  for (int mi = 0; mi < MT; ++mi) {
#pragma unroll
    for (int j = 0; j < NT; ++j) {
      const int n = n0b + wn + j * 16 + col;
      const float bb = (n < nvalid) ? bias[n] : 0.0f;
#pragma unroll
      for (int i = 0; i < 8; ++i) {
        const int m = m0b + wm + mi * 16 + rb + i;
        const float val = acc[mi][j][i] + bb;
        if (E == 0) {
          if (n < N) ((float*)D)[(size_t)m * ldd + n] = val;
        } else if (E == 1) {
          if (n < N) ((__bf16*)D)[(size_t)m * ldd + n] = (__bf16)fmaxf(val, 0.0f);
        } else {
          if (n < nvalid) ((float*)D)[(size_t)m * ldd + n] = sigmoidf_(val);
        }
      }
    }
  }
}

// ---- one-time weight prep ----
__global__ void k_f32_to_bf16(const float* __restrict__ s,
                              __bf16* __restrict__ d, int n) {
  for (int i = blockIdx.x * blockDim.x + threadIdx.x; i < n;
       i += gridDim.x * blockDim.x)
    d[i] = (__bf16)s[i];
}

__global__ void k_build_wg(const float* __restrict__ wih,
                           const float* __restrict__ whh,
                           __bf16* __restrict__ wg) {
  // wg: [1024][512] = [Wih | Whh]
  int i = blockIdx.x * blockDim.x + threadIdx.x;
  if (i >= kG4 * kGK) return;
  int n = i >> 9, k = i & (kGK - 1);
  float v = (k < kHID) ? wih[n * kINS + k] : whh[n * kHID + (k - kHID)];
  wg[i] = (__bf16)v;
}

__global__ void k_build_wv(const float* __restrict__ wv,
                           __bf16* __restrict__ wvp) {
  // wvp: [16][CAT], rows >= ND zero-padded
  int i = blockIdx.x * blockDim.x + threadIdx.x;
  if (i >= 16 * kCAT) return;
  int n = i / kCAT, k = i - n * kCAT;
  wvp[i] = (__bf16)((n < kND) ? wv[(size_t)n * kCAT + k] : 0.0f);
}

__global__ void k_gbias(const float* __restrict__ bih,
                        const float* __restrict__ bhh,
                        float* __restrict__ gb) {
  int i = blockIdx.x * blockDim.x + threadIdx.x;
  if (i < kG4) gb[i] = bih[i] + bhh[i];
}

__global__ void k_init(float* __restrict__ cbuf, __bf16* __restrict__ gin) {
  int i = blockIdx.x * blockDim.x + threadIdx.x;
  if (i < kBS * kHID) cbuf[i] = 0.0f;
  if (i < kBS * kGK)  gin[i]  = (__bf16)0.0f;
}

// ---- per-step small kernels ----
__global__ void k_embed(const float* __restrict__ desc,
                        const float* __restrict__ Wd,
                        const float* __restrict__ bd,
                        __bf16* __restrict__ xb, int t) {
  int idx = blockIdx.x * blockDim.x + threadIdx.x;  // BS*ND*INS
  if (idx >= kBS * kND * kINS) return;
  int i = idx & (kINS - 1);
  int n = (idx >> 8) % kND;
  int b = idx / (kND * kINS);
  const float* dp = desc + (((size_t)b * kT + t) * kND + n) * kDS;
  const float* wp = Wd + i * kDS;
  float e = bd[i];
#pragma unroll
  for (int d = 0; d < kDS; ++d) e += dp[d] * wp[d];
  xb[(size_t)b * kCAT + n * kINS + i] = (__bf16)e;
}

__global__ void k_lstm(const float* __restrict__ gates,
                       float* __restrict__ cbuf,
                       __bf16* __restrict__ xb,
                       __bf16* __restrict__ gin) {
  int idx = blockIdx.x * blockDim.x + threadIdx.x;  // BS*HID
  if (idx >= kBS * kHID) return;
  int b = idx >> 8, j = idx & (kHID - 1);
  const float* g = gates + (size_t)b * kG4;
  float ig = sigmoidf_(g[j]);
  float fg = sigmoidf_(g[kHID + j]);
  float gg = tanhf(g[2 * kHID + j]);
  float og = sigmoidf_(g[3 * kHID + j]);
  float c = fg * cbuf[idx] + ig * gg;
  cbuf[idx] = c;
  float h = og * tanhf(c);
  __bf16 hb = (__bf16)h;
  xb[(size_t)b * kCAT + (kND * kINS) + j] = hb;   // x = [emb | h]
  gin[(size_t)b * kGK + kHID + j]        = hb;   // gates input = [inp | h]
}

extern "C" void kernel_launch(void* const* d_in, const int* in_sizes, int n_in,
                              void* d_out, int out_size, void* d_ws, size_t ws_size,
                              hipStream_t stream) {
  (void)in_sizes; (void)n_in; (void)out_size; (void)ws_size;
  const float* descriptors = (const float*)d_in[0];
  const float* Wd  = (const float*)d_in[1];
  const float* bd  = (const float*)d_in[2];
  const float* W1  = (const float*)d_in[3];
  const float* b1  = (const float*)d_in[4];
  const float* W2  = (const float*)d_in[5];
  const float* b2  = (const float*)d_in[6];
  const float* Wv  = (const float*)d_in[7];
  const float* bv  = (const float*)d_in[8];
  const float* Wih = (const float*)d_in[9];
  const float* Whh = (const float*)d_in[10];
  const float* bih = (const float*)d_in[11];
  const float* bhh = (const float*)d_in[12];
  float* out = (float*)d_out;

  // workspace carve (~27.4 MB, 256B aligned chunks)
  char* p = (char*)d_ws;
  auto take = [&](size_t bytes) -> char* {
    char* r = p;
    p += (bytes + 255) & ~(size_t)255;
    return r;
  };
  __bf16* W1b   = (__bf16*)take((size_t)kCAT * kCAT * 2);
  __bf16* W2b   = (__bf16*)take((size_t)kHID * kCAT * 2);
  __bf16* WvPb  = (__bf16*)take((size_t)16 * kCAT * 2);
  __bf16* Wgb   = (__bf16*)take((size_t)kG4 * kGK * 2);
  float*  gbias = (float*)take((size_t)kG4 * 4);
  __bf16* xb    = (__bf16*)take((size_t)kBS * kCAT * 2);
  __bf16* zb    = (__bf16*)take((size_t)kBS * kCAT * 2);
  __bf16* gin   = (__bf16*)take((size_t)kBS * kGK * 2);
  float*  gates = (float*)take((size_t)kBS * kG4 * 4);
  float*  cbuf  = (float*)take((size_t)kBS * kHID * 4);

  const dim3 blk(256);

  // one-time prep
  k_f32_to_bf16<<<2048, blk, 0, stream>>>(W1, W1b, kCAT * kCAT);
  k_f32_to_bf16<<<512,  blk, 0, stream>>>(W2, W2b, kHID * kCAT);
  k_build_wg<<<(kG4 * kGK + 255) / 256, blk, 0, stream>>>(Wih, Whh, Wgb);
  k_build_wv<<<(16 * kCAT + 255) / 256, blk, 0, stream>>>(Wv, WvPb);
  k_gbias<<<(kG4 + 255) / 256, blk, 0, stream>>>(bih, bhh, gbias);
  k_init<<<(kBS * kGK + 255) / 256, blk, 0, stream>>>(cbuf, gin);

  for (int t = 0; t < kT; ++t) {
    // gates = gin @ Wg^T + (bih+bhh)   M=512 N=1024 K=512 -> f32
    gemm_bf16_wmma<0><<<dim3(kBS / 128, kG4 / 128), blk, 0, stream>>>(
        gin, kGK, Wgb, kGK, gbias, gates, kG4, kBS, kG4, kGK, kG4);
    // LSTM pointwise: c,h update; h -> xb[:,2560:], gin[:,256:]
    k_lstm<<<(kBS * kHID + 255) / 256, blk, 0, stream>>>(gates, cbuf, xb, gin);
    // emb_t -> xb[:, :2560]
    k_embed<<<(kBS * kND * kINS + 255) / 256, blk, 0, stream>>>(
        descriptors, Wd, bd, xb, t);
    // z = relu(x @ W1^T + b1)   M=512 N=2816 K=2816 -> bf16
    gemm_bf16_wmma<1><<<dim3(kBS / 128, kCAT / 128), blk, 0, stream>>>(
        xb, kCAT, W1b, kCAT, b1, zb, kCAT, kBS, kCAT, kCAT, kCAT);
    // nxt = relu(z @ W2^T + b2) -> bf16 into gin[:, :256]
    gemm_bf16_wmma<1><<<dim3(kBS / 128, kHID / 128), blk, 0, stream>>>(
        zb, kCAT, W2b, kCAT, b2, gin, kGK, kBS, kHID, kCAT, kHID);
    // att = sigmoid(z @ Wv^T + bv) -> out[:, t, :]  (N padded to 16, cols<10)
    gemm_bf16_wmma<2><<<dim3(kBS / 128, 1), blk, 0, stream>>>(
        zb, kCAT, WvPb, kCAT, bv, out + (size_t)t * kND, kT * kND,
        kBS, 16, kCAT, kND);
  }
}
